// HQSAAttention_63754494541993
// MI455X (gfx1250) — matrix operand order repaired and checked by reading, not compiled, not run on hardware
//
#include <hip/hip_runtime.h>
#include <hip/hip_bf16.h>

typedef _Float16 v16h __attribute__((ext_vector_type(16)));
typedef _Float16 v8h  __attribute__((ext_vector_type(8)));
typedef float    v8f  __attribute__((ext_vector_type(8)));

static constexpr int DM  = 1024;  // d_model
static constexpr int H   = 16;    // heads
static constexpr int NB  = 64;    // bins
static constexpr int DH  = 64;    // d_head
static constexpr int SEQ = 4096;  // sequence length

static __device__ __forceinline__ v16h cat8(v8h lo, v8h hi) {
  return __builtin_shufflevector(lo, hi, 0,1,2,3,4,5,6,7,8,9,10,11,12,13,14,15);
}

static __device__ __forceinline__ v8f wmma_f16(v16h a, v16h b, v8f c) {
  // D(f32 16x16) = A(f16 16x32) * B(f16 32x16) + C
  return __builtin_amdgcn_wmma_f32_16x16x32_f16(false, a, false, b, (short)0, c,
                                                false, false);
}

// Async copy 16 B from global to LDS (per lane), tracked by ASYNCcnt.
static __device__ __forceinline__ void async_b128(unsigned lds_off, const void* gaddr) {
  asm volatile("global_load_async_to_lds_b128 %0, %1, off"
               :: "v"(lds_off), "v"(gaddr)
               : "memory");
}

static __device__ __forceinline__ void wait_async0() {
#if __has_builtin(__builtin_amdgcn_s_wait_asynccnt)
  __builtin_amdgcn_s_wait_asynccnt(0);
#else
  asm volatile("s_wait_asynccnt 0x0" ::: "memory");
#endif
}

// ---------------------------------------------------------------- prep kernels

__global__ __launch_bounds__(256) void k_f32_to_f16(const float* __restrict__ in,
                                                    _Float16* __restrict__ out,
                                                    long n) {
  long i = ((long)blockIdx.x * 256 + threadIdx.x) * 4;
  if (i + 3 < n) {
    float4 v = *(const float4*)(in + i);
    out[i + 0] = (_Float16)v.x;
    out[i + 1] = (_Float16)v.y;
    out[i + 2] = (_Float16)v.z;
    out[i + 3] = (_Float16)v.w;
  }
}

// WT[k, n] = (f16) W[n, k]   (W is [N, K] row-major; result is B = W^T [K, N])
__global__ __launch_bounds__(256) void k_wtrans(const float* __restrict__ W,
                                                _Float16* __restrict__ WT) {
  int idx = blockIdx.x * 256 + threadIdx.x;  // over DM*DM, enumerated k*DM + n
  int k = idx >> 10, n = idx & (DM - 1);
  WT[(size_t)k * DM + n] = (_Float16)W[(size_t)n * DM + k];
}

// M[h, d, n] = sum_l W_bin[l, d] * proto[h, n, l]  (collapses z->logits chain)
__global__ __launch_bounds__(256) void k_build_m(const float* __restrict__ Wbin,
                                                 const float* __restrict__ proto,
                                                 float* __restrict__ Mm) {
  int idx = blockIdx.x * 256 + threadIdx.x;  // H*64*64
  int n = idx & 63, d = (idx >> 6) & 63, h = idx >> 12;
  float acc = 0.f;
#pragma unroll
  for (int l = 0; l < 8; ++l)
    acc += Wbin[l * DH + d] * proto[((size_t)h * NB + n) * 8 + l];
  Mm[idx] = acc;
}

// ------------------------------------------------------------- generic WMMA GEMM
// C[M,N] = A[M,K] @ B[K,N] + bias.
// Block = 8 waves, tile 256(M) x 64(N); each wave computes 32x64 (2 A-frags
// sharing 4 B-frags). B tile (32x64 f16 = 4 KB) is staged through LDS with
// double-buffered global_load_async_to_lds_b128 (ASYNCcnt).
__global__ __launch_bounds__(256) void k_gemm_f16(
    const _Float16* __restrict__ A, const _Float16* __restrict__ B,
    const float* __restrict__ bias, float* __restrict__ Cf,
    _Float16* __restrict__ Ch, int M, int N, int K) {
  __shared__ __align__(32) _Float16 bts[2][32 * 64];  // 8 KB double buffer
  const int lane = threadIdx.x & 31;
  const int wv = threadIdx.x >> 5;
  const int ln = lane & 15, hl = lane >> 4;
  const int mBlocks = M >> 8;
  const int m0 = (blockIdx.x % mBlocks) * 256 + wv * 32;
  const int n0 = (blockIdx.x / mBlocks) * 64;

  // Cooperative async staging: 256 threads x 16 B = one 4 KB B tile.
  const int tRow = threadIdx.x >> 3;         // 0..31 (k within tile)
  const int tCol = (threadIdx.x & 7) * 8;    // f16 column chunk
  const _Float16* gB = B + (size_t)tRow * N + n0 + tCol;
  const unsigned lds0 = (unsigned)(uintptr_t)&bts[0][tRow * 64 + tCol];
  const unsigned lds1 = (unsigned)(uintptr_t)&bts[1][tRow * 64 + tCol];

  const _Float16* aRow0 = A + (size_t)(m0 + ln) * K;        // A-frag rows m0..m0+15
  const _Float16* aRow1 = A + (size_t)(m0 + 16 + ln) * K;   // A-frag rows +16

  async_b128(lds0, gB);                       // prologue: tile k0 = 0 -> buf 0

  v8f acc0[4] = {}, acc1[4] = {};
  for (int k0 = 0; k0 < K; k0 += 32) {
    const int p = (k0 >> 5) & 1;
    wait_async0();                            // my chunks of buf p landed
    __syncthreads();                          // everyone's chunks landed
    if (k0 + 32 < K)                          // prefetch next tile -> buf p^1
      async_b128(p ? lds0 : lds1, gB + (size_t)(k0 + 32) * N);

    __builtin_prefetch(aRow0 + k0 + 128, 0, 3);
    v16h a0 = cat8(*(const v8h*)(aRow0 + k0 + 8 * hl),
                   *(const v8h*)(aRow0 + k0 + 16 + 8 * hl));
    v16h a1 = cat8(*(const v8h*)(aRow1 + k0 + 8 * hl),
                   *(const v8h*)(aRow1 + k0 + 16 + 8 * hl));
    const _Float16* bl = &bts[p][ln];  // B-frag: column n per lane, k in table order
#pragma unroll
    for (int f = 0; f < 4; ++f) {
      v16h bf;
#pragma unroll
      for (int s = 0; s < 16; ++s) bf[s] = bl[(8 * hl + (s & 7) + ((s >> 3) << 4)) * 64 + 16 * f];
      acc0[f] = wmma_f16(a0, bf, acc0[f]);
      acc1[f] = wmma_f16(a1, bf, acc1[f]);
    }
  }
#pragma unroll
  for (int f = 0; f < 4; ++f) {
    const int n = n0 + 16 * f + ln;
    const float bv = bias[n];
#pragma unroll
    for (int v = 0; v < 8; ++v) {
      const int ma = m0 + 8 * hl + v;
      const int mb = ma + 16;
      const float va = acc0[f][v] + bv;
      const float vb = acc1[f][v] + bv;
      if (Cf) { Cf[(size_t)ma * N + n] = va; Cf[(size_t)mb * N + n] = vb; }
      if (Ch) { Ch[(size_t)ma * N + n] = (_Float16)va; Ch[(size_t)mb * N + n] = (_Float16)vb; }
    }
  }
}

// ------------------------------------------------- bin softmax (probs, transposed)
// One wave per (b,h,s); logits_n = (K_row . M_h[:,n]) / tau; softmax over 64 bins.
__global__ __launch_bounds__(256) void k_bin_probs(
    const _Float16* __restrict__ Kh,      // [B*S, DM] (b,s,h,d)
    const float* __restrict__ Mm,         // [H, 64, 64]
    const float* __restrict__ log_tau,    // [H]
    _Float16* __restrict__ probsT) {      // [B, H, NB, S]
  const int lane = threadIdx.x & 31;
  const int wv = threadIdx.x >> 5;
  const long row = (long)blockIdx.x * 8 + wv;  // over B*H*S
  const int s = (int)(row & (SEQ - 1));
  const int bh = (int)(row >> 12);             // SEQ == 4096
  const int h = bh & (H - 1);
  const int b = bh >> 4;

  const _Float16* kp = Kh + ((size_t)b * SEQ + s) * DM + h * DH;
  const float kA = (float)kp[2 * lane];
  const float kB = (float)kp[2 * lane + 1];
  const float* Mh = Mm + (size_t)h * 64 * 64;

  float a0 = 0.f, a1 = 0.f;
#pragma unroll
  for (int d = 0; d < 64; ++d) {
    const float kd = __shfl((d & 1) ? kB : kA, d >> 1, 32);
    a0 += kd * Mh[d * 64 + lane];
    a1 += kd * Mh[d * 64 + lane + 32];
  }
  const float tau = fmaxf(__expf(log_tau[h]), 1e-3f) + 1e-8f;
  a0 /= tau; a1 /= tau;
  float mx = fmaxf(a0, a1);
#pragma unroll
  for (int m = 16; m >= 1; m >>= 1) mx = fmaxf(mx, __shfl_xor(mx, m, 32));
  const float e0 = __expf(a0 - mx), e1 = __expf(a1 - mx);
  float sm = e0 + e1;
#pragma unroll
  for (int m = 16; m >= 1; m >>= 1) sm += __shfl_xor(sm, m, 32);
  const float inv = 1.f / sm;
  _Float16* pr = probsT + ((size_t)bh * NB) * SEQ + s;
  pr[(size_t)lane * SEQ] = (_Float16)(e0 * inv);
  pr[(size_t)(lane + 32) * SEQ] = (_Float16)(e1 * inv);
}

// ------------------------------------------- soft aggregation into bins (per b,h)
// weighted_K = P^T @ K, weighted_V = P^T @ V (64 x 4096 x 64 WMMA GEMMs),
// counts from the A fragment; then the faithful divide-by-count[d] broadcast.
__global__ __launch_bounds__(128) void k_aggregate(
    const _Float16* __restrict__ probsT,  // [B,H,NB,S]
    const _Float16* __restrict__ Kh,      // [B*S, DM]
    const _Float16* __restrict__ Vh,      // [B*S, DM]
    _Float16* __restrict__ KbT,           // [B,H, DH(d), NB(n)]  (transposed)
    _Float16* __restrict__ Vb) {          // [B,H, NB(n), DH(d)]
  __shared__ float cnts[64];
  const int lane = threadIdx.x & 31;
  const int wv = threadIdx.x >> 5;
  const int ln = lane & 15, hl = lane >> 4;
  const int bh = blockIdx.x;
  const int h = bh & (H - 1), b = bh >> 4;

  const _Float16* pRow = probsT + ((size_t)bh * NB + wv * 16 + ln) * SEQ;
  const _Float16* kBase = Kh + ((size_t)b * SEQ) * DM + h * DH + ln;
  const _Float16* vBase = Vh + ((size_t)b * SEQ) * DM + h * DH + ln;

  v8f kacc[4] = {}, vacc[4] = {};
  float cpart = 0.f;
  for (int s0 = 0; s0 < SEQ; s0 += 32) {
    v8h plo = *(const v8h*)(pRow + s0 + 8 * hl);
    v8h phi = *(const v8h*)(pRow + s0 + 16 + 8 * hl);
    v16h a = cat8(plo, phi);
#pragma unroll
    for (int i = 0; i < 8; ++i) cpart += (float)plo[i] + (float)phi[i];
    const _Float16* kr = kBase + (size_t)s0 * DM;
    const _Float16* vr = vBase + (size_t)s0 * DM;
#pragma unroll
    for (int f = 0; f < 4; ++f) {
      v16h kf, vf;
#pragma unroll
      for (int s = 0; s < 16; ++s) {
        const size_t o = (size_t)(8 * hl + (s & 7) + ((s >> 3) << 4)) * DM + 16 * f;
        kf[s] = kr[o]; vf[s] = vr[o];
      }
      kacc[f] = wmma_f16(a, kf, kacc[f]);
      vacc[f] = wmma_f16(a, vf, vacc[f]);
    }
  }
  cpart += __shfl_xor(cpart, 16, 32);        // combine the two K-halves of row n
  if (lane < 16) cnts[wv * 16 + lane] = cpart;
  __syncthreads();
#pragma unroll
  for (int f = 0; f < 4; ++f) {
    const float inv = 1.f / (cnts[16 * f + ln] + 1e-6f);  // divisor indexed by d!
#pragma unroll
    for (int v = 0; v < 8; ++v) {
      const int nrow = wv * 16 + 8 * hl + v;  // bin n
      const int dcol = 16 * f + ln;           // feature d
      Vb[((size_t)bh * NB + nrow) * DH + dcol] = (_Float16)(vacc[f][v] * inv);
      KbT[((size_t)bh * DH + dcol) * NB + nrow] = (_Float16)(kacc[f][v] * inv);
    }
  }
}

// -------------------------------------------- attention over bins (fused, per tile)
// scores = Q @ Kb^T / 8 -> row softmax -> context = W @ Vb; 16 s-rows per wave.
__global__ __launch_bounds__(256) void k_attn(
    const _Float16* __restrict__ Qh,   // [B*S, DM]
    const _Float16* __restrict__ KbT,  // [B,H, DH, NB]
    const _Float16* __restrict__ Vb,   // [B,H, NB, DH]
    _Float16* __restrict__ ctx) {      // [B*S, DM] (b,s,h,d)
  __shared__ __align__(32) _Float16 wtile[8 * 16 * 64];
  const int lane = threadIdx.x & 31;
  const int wv = threadIdx.x >> 5;
  const int ln = lane & 15, hl = lane >> 4;
  const int tilesPerBH = SEQ / 128;
  const int bh = blockIdx.x / tilesPerBH;
  const int st = blockIdx.x % tilesPerBH;
  const int h = bh & (H - 1), b = bh >> 4;
  const int s0 = st * 128 + wv * 16;

  const _Float16* qRow = Qh + ((size_t)b * SEQ + s0 + ln) * DM + h * DH;
  v16h qa0 = cat8(*(const v8h*)(qRow + 8 * hl), *(const v8h*)(qRow + 16 + 8 * hl));
  v16h qa1 = cat8(*(const v8h*)(qRow + 32 + 8 * hl), *(const v8h*)(qRow + 48 + 8 * hl));

  const _Float16* kb0 = KbT + ((size_t)bh * DH) * NB + ln;  // d = 0..31
  const _Float16* kb1 = kb0 + 32 * NB;                                // d = 32..63
  v8f sc[4] = {};
#pragma unroll
  for (int f = 0; f < 4; ++f) {
    v16h kf0, kf1;
#pragma unroll
    for (int s = 0; s < 16; ++s) {
      const int o = (8 * hl + (s & 7) + ((s >> 3) << 4)) * NB + 16 * f;
      kf0[s] = kb0[o]; kf1[s] = kb1[o];
    }
    sc[f] = wmma_f16(qa0, kf0, sc[f]);
    sc[f] = wmma_f16(qa1, kf1, sc[f]);
  }

  _Float16* wt = wtile + wv * (16 * 64);
#pragma unroll
  for (int v = 0; v < 8; ++v) {
    float x0 = sc[0][v] * 0.125f, x1 = sc[1][v] * 0.125f;
    float x2 = sc[2][v] * 0.125f, x3 = sc[3][v] * 0.125f;
    float mx = fmaxf(fmaxf(x0, x1), fmaxf(x2, x3));
#pragma unroll
    for (int m = 8; m >= 1; m >>= 1) mx = fmaxf(mx, __shfl_xor(mx, m, 32));
    float e0 = __expf(x0 - mx), e1 = __expf(x1 - mx);
    float e2 = __expf(x2 - mx), e3 = __expf(x3 - mx);
    float sm = e0 + e1 + e2 + e3;
#pragma unroll
    for (int m = 8; m >= 1; m >>= 1) sm += __shfl_xor(sm, m, 32);
    const float inv = 1.f / sm;
    const int r = 8 * hl + v;
    wt[r * 64 + 0 + ln] = (_Float16)(e0 * inv);
    wt[r * 64 + 16 + ln] = (_Float16)(e1 * inv);
    wt[r * 64 + 32 + ln] = (_Float16)(e2 * inv);
    wt[r * 64 + 48 + ln] = (_Float16)(e3 * inv);
  }
  __syncthreads();

  const _Float16* wr = wt + ln * 64;   // re-stage weights in A-fragment layout
  v16h wa0 = cat8(*(const v8h*)(wr + 8 * hl), *(const v8h*)(wr + 16 + 8 * hl));
  v16h wa1 = cat8(*(const v8h*)(wr + 32 + 8 * hl), *(const v8h*)(wr + 48 + 8 * hl));

  const _Float16* vb0 = Vb + ((size_t)bh * NB) * DH + ln;  // n = 0..31
  const _Float16* vb1 = vb0 + 32 * DH;                               // n = 32..63
  v8f cacc[4] = {};
#pragma unroll
  for (int f = 0; f < 4; ++f) {
    v16h vf0, vf1;
#pragma unroll
    for (int s = 0; s < 16; ++s) {
      const int o = (8 * hl + (s & 7) + ((s >> 3) << 4)) * DH + 16 * f;
      vf0[s] = vb0[o]; vf1[s] = vb1[o];
    }
    cacc[f] = wmma_f16(wa0, vf0, cacc[f]);
    cacc[f] = wmma_f16(wa1, vf1, cacc[f]);
  }
#pragma unroll
  for (int f = 0; f < 4; ++f)
#pragma unroll
    for (int v = 0; v < 8; ++v) {
      const int m = s0 + 8 * hl + v;
      ctx[((size_t)b * SEQ + m) * DM + h * DH + 16 * f + ln] = (_Float16)cacc[f][v];
    }
}

// ------------------------------------------------------------------------- launch

extern "C" void kernel_launch(void* const* d_in, const int* in_sizes, int n_in,
                              void* d_out, int out_size, void* d_ws, size_t ws_size,
                              hipStream_t stream) {
  const float* x    = (const float*)d_in[0];
  const float* Wq   = (const float*)d_in[1];
  const float* bq   = (const float*)d_in[2];
  const float* Wk   = (const float*)d_in[3];
  const float* bk   = (const float*)d_in[4];
  const float* Wv   = (const float*)d_in[5];
  const float* bv   = (const float*)d_in[6];
  const float* Wo   = (const float*)d_in[7];
  const float* bo   = (const float*)d_in[8];
  const float* Wbin = (const float*)d_in[9];
  const float* prot = (const float*)d_in[10];
  const float* ltau = (const float*)d_in[11];
  (void)n_in; (void)out_size; (void)ws_size;

  const int Bb = in_sizes[0] / (SEQ * DM);   // batch = 4
  const long Mrows = (long)Bb * SEQ;         // 16384

  char* ws = (char*)d_ws;
  size_t off = 0;
  auto carve = [&](size_t bytes) -> char* {
    char* p = ws + off;
    off = (off + bytes + 255) & ~((size_t)255);
    return p;
  };
  _Float16* WTq = (_Float16*)carve((size_t)DM * DM * 2);
  _Float16* WTk = (_Float16*)carve((size_t)DM * DM * 2);
  _Float16* WTv = (_Float16*)carve((size_t)DM * DM * 2);
  _Float16* WTo = (_Float16*)carve((size_t)DM * DM * 2);
  float*    Mm  = (float*)carve((size_t)H * 64 * 64 * sizeof(float));
  _Float16* Qh  = (_Float16*)carve((size_t)Mrows * DM * 2);
  _Float16* Kh  = (_Float16*)carve((size_t)Mrows * DM * 2);
  _Float16* Vh  = (_Float16*)carve((size_t)Mrows * DM * 2);
  _Float16* Xh  = (_Float16*)carve((size_t)Mrows * DM * 2);  // x_h, later reused as ctx
  _Float16* Pt  = (_Float16*)carve((size_t)Bb * H * NB * SEQ * 2);
  _Float16* KbT = (_Float16*)carve((size_t)Bb * H * DH * NB * 2);
  _Float16* Vb  = (_Float16*)carve((size_t)Bb * H * NB * DH * 2);

  const long nx = Mrows * DM;
  k_f32_to_f16<<<dim3((unsigned)((nx / 4 + 255) / 256)), dim3(256), 0, stream>>>(x, Xh, nx);

  const unsigned wblk = (DM * DM) / 256;
  k_wtrans<<<dim3(wblk), dim3(256), 0, stream>>>(Wq, WTq);
  k_wtrans<<<dim3(wblk), dim3(256), 0, stream>>>(Wk, WTk);
  k_wtrans<<<dim3(wblk), dim3(256), 0, stream>>>(Wv, WTv);
  k_wtrans<<<dim3(wblk), dim3(256), 0, stream>>>(Wo, WTo);
  k_build_m<<<dim3((H * 64 * 64) / 256), dim3(256), 0, stream>>>(Wbin, prot, Mm);

  const unsigned ggrid = (unsigned)((Mrows >> 8) * (DM / 64));   // 256-row M tiles
  k_gemm_f16<<<dim3(ggrid), dim3(256), 0, stream>>>(Xh, WTq, bq, (float*)nullptr, Qh,
                                                    (int)Mrows, DM, DM);
  k_gemm_f16<<<dim3(ggrid), dim3(256), 0, stream>>>(Xh, WTk, bk, (float*)nullptr, Kh,
                                                    (int)Mrows, DM, DM);
  k_gemm_f16<<<dim3(ggrid), dim3(256), 0, stream>>>(Xh, WTv, bv, (float*)nullptr, Vh,
                                                    (int)Mrows, DM, DM);

  k_bin_probs<<<dim3((unsigned)(Mrows * H / 8)), dim3(256), 0, stream>>>(Kh, Mm, ltau, Pt);
  k_aggregate<<<dim3((unsigned)(Bb * H)), dim3(128), 0, stream>>>(Pt, Kh, Vh, KbT, Vb);
  k_attn<<<dim3((unsigned)(Bb * H * (SEQ / 128))), dim3(256), 0, stream>>>(Qh, KbT, Vb, Xh);

  k_gemm_f16<<<dim3(ggrid), dim3(256), 0, stream>>>(Xh, WTo, bo, (float*)d_out,
                                                    (_Float16*)nullptr, (int)Mrows, DM, DM);
}
